// PointTransformer_55173149884583
// MI455X (gfx1250) — compile-verified, matrix operand order checked by reading, not run on hardware
//
#include <hip/hip_runtime.h>
#include <hip/hip_bf16.h>
#include <math.h>

#define PB 32
#define PN 8192
#define PG 256
#define PK 32
#define PT 257
#define TPAD 288
#define EMB 384
#define NH 6
#define HDIM 64
#define NLAYER 12
#define ENCD 256
#define LEPS 1e-5f

typedef __attribute__((ext_vector_type(16))) __bf16 v16bf;
typedef __attribute__((ext_vector_type(8)))  float  v8f;

__device__ __forceinline__ float  b2f(__bf16 x) { return (float)x; }
__device__ __forceinline__ __bf16 f2b(float x)  { return (__bf16)x; }
__device__ __forceinline__ float  gelu_f(float v) {
  return 0.5f * v * (1.f + erff(v * 0.70710678118654752f));
}

// ---------------------------------------------------------------------------
// WMMA GEMM: D = A(bf16, MxK) x B^T(bf16, NxK row-major) [+bias][+act][epilogue]
// One wave computes a 16x(16*NT) tile; K-loop in steps of 32
// (v_wmma_f32_16x16x32_bf16). All operand loads are contiguous 2x16B per
// fragment, so they lower to global_load_b128 pairs.
//
// Template params:
//   AMODE: 0 = row-major A, 1 = concat [fg(group) | A] (ENCD split)
//   EPI:   0 f32, 1 bf16, 2 qkv-scatter, 3 attn-gather, 4 rowshift, 5 residual
//   ACT:   0 none, 1 relu, 2 gelu
//   NT:    column tiles per wave (N must be divisible by 16*NT)
// ---------------------------------------------------------------------------
struct GemmP {
  const __bf16* A; const __bf16* B; const float* bias;
  void* out; const float* R1; const float* R2; const __bf16* Afg;
  __bf16 *q, *k2, *v;
  long long strideA, strideB, strideC;
  int M, N, K, lda, ldb, ldc;
};

template <int AMODE, int EPI, int ACT, int NT>
__global__ __launch_bounds__(256) void k_gemm(GemmP p) {
  const int z = blockIdx.y;
  const __bf16* Ab = p.A + (long long)z * p.strideA;
  const __bf16* Bb = p.B + (long long)z * p.strideB;
  const int lane = threadIdx.x & 31;
  const int wv   = threadIdx.x >> 5;
  const int tilesN = p.N / (16 * NT);
  const int tile = blockIdx.x * 8 + wv;
  if (tile >= (p.M >> 4) * tilesN) return;
  const int m0 = (tile / tilesN) << 4;
  const int n0 = (tile % tilesN) * (16 * NT);
  const int hl  = lane >> 4;
  const int l16 = lane & 15;
  const int arow = m0 + l16;
  v8f acc[NT];
#pragma unroll
  for (int j = 0; j < NT; ++j) acc[j] = (v8f){};
  for (int k0 = 0; k0 < p.K; k0 += 32) {
    v16bf a = {};
    if (AMODE == 0) {
      const __bf16* ap = Ab + (long long)arow * p.lda + (k0 + 8 * hl);
      __builtin_prefetch(ap + 64, 0, 1);
#pragma unroll
      for (int i = 0; i < 8; ++i) a[i] = ap[i];
#pragma unroll
      for (int i = 0; i < 8; ++i) a[8 + i] = ap[16 + i];
    } else {
      const int grp = arow >> 5;  // PK=32 rows per group
#pragma unroll
      for (int i = 0; i < 16; ++i) {
        int kk = k0 + ((i < 8) ? i : i + 8) + 8 * hl;
        a[i] = (kk < ENCD) ? p.Afg[(long long)grp * ENCD + kk]
                           : Ab[(long long)arow * ENCD + (kk - ENCD)];
      }
    }
#pragma unroll
    for (int j = 0; j < NT; ++j) {
      v16bf b = {};
      const __bf16* bp = Bb + (long long)(n0 + j * 16 + l16) * p.ldb + (k0 + 8 * hl);
#pragma unroll
      for (int i = 0; i < 8; ++i) b[i] = bp[i];
#pragma unroll
      for (int i = 0; i < 8; ++i) b[8 + i] = bp[16 + i];
      acc[j] = __builtin_amdgcn_wmma_f32_16x16x32_bf16(false, a, false, b,
                                                       (short)0, acc[j], false, false);
    }
  }
#pragma unroll
  for (int j = 0; j < NT; ++j) {
#pragma unroll
    for (int r = 0; r < 8; ++r) {
      const int mr = m0 + r + 8 * hl;
      const int nc = n0 + j * 16 + l16;
      float vv = acc[j][r];
      if (p.bias) vv += p.bias[nc];
      if (ACT == 1)      vv = fmaxf(vv, 0.f);
      else if (ACT == 2) vv = gelu_f(vv);
      if (EPI == 0) {
        ((float*)p.out)[(long long)z * p.strideC + (long long)mr * p.ldc + nc] = vv;
      } else if (EPI == 1) {
        ((__bf16*)p.out)[(long long)z * p.strideC + (long long)mr * p.ldc + nc] = f2b(vv);
      } else if (EPI == 2) {           // qkv scatter; V stored transposed (d x t)
        const int bb = mr / PT, t = mr - bb * PT;
        const int three = nc / EMB, rem = nc - three * EMB;
        const int hh = rem >> 6, d = rem & 63;
        if (three == 2) {
          p.v[(((long long)(bb * NH + hh)) * HDIM + d) * TPAD + t] = f2b(vv);
        } else {
          __bf16* dst = (three == 0) ? p.q : p.k2;
          dst[(((long long)(bb * NH + hh)) * TPAD + t) * HDIM + d] = f2b(vv);
        }
      } else if (EPI == 3) {           // attention output gather (b,t,h*64+d)
        const int bb = z / NH, hh = z - bb * NH;
        if (mr < PT)
          ((__bf16*)p.out)[((long long)(bb * PT + mr)) * EMB + hh * HDIM + nc] = f2b(vv);
      } else if (EPI == 4) {           // row-shift: (b,g) -> token slot (b, g+1)
        const int bb = mr >> 8, g = mr & 255;
        ((float*)p.out)[((long long)(bb * PT + g + 1)) * EMB + nc] = vv;
      } else {                         // residual f32: acc+bias+R1(+R2)
        const long long ix = (long long)mr * p.ldc + nc;
        float o = vv + p.R1[ix];
        if (p.R2) o += p.R2[ix];
        ((float*)p.out)[ix] = o;
      }
    }
  }
}

// ---------------------------------------------------------------------------
// Farthest point sampling: one block per batch
// ---------------------------------------------------------------------------
__global__ __launch_bounds__(256) void k_fps(const float* pts, int* idx) {
  const int b = blockIdx.x;
  const float* P = pts + (long long)b * PN * 3;
  __shared__ float sv[256];
  __shared__ int   si[256];
  __shared__ float cx, cy, cz;
  __shared__ int   sFar;
  const int tid = threadIdx.x;
  float dist[32];
#pragma unroll
  for (int j = 0; j < 32; ++j) dist[j] = 1e10f;
  if (tid == 0) sFar = 0;
  __syncthreads();
  for (int it = 0; it < PG; ++it) {
    const int far = sFar;
    if (tid == 0) {
      idx[b * PG + it] = far;
      cx = P[far * 3]; cy = P[far * 3 + 1]; cz = P[far * 3 + 2];
    }
    __syncthreads();
    const float fx = cx, fy = cy, fz = cz;
    float bv = -1.f; int bi = 0;
    for (int j = 0; j < 32; ++j) {
      const int pI = tid + 256 * j;
      const float dx = P[pI * 3] - fx, dy = P[pI * 3 + 1] - fy, dz = P[pI * 3 + 2] - fz;
      float nd = fminf(dist[j], dx * dx + dy * dy + dz * dz);
      dist[j] = nd;
      if (nd > bv) { bv = nd; bi = pI; }
    }
    sv[tid] = bv; si[tid] = bi;
    __syncthreads();
    for (int st = 128; st > 0; st >>= 1) {
      if (tid < st) {
        if (sv[tid + st] > sv[tid] ||
            (sv[tid + st] == sv[tid] && si[tid + st] < si[tid])) {
          sv[tid] = sv[tid + st]; si[tid] = si[tid + st];
        }
      }
      __syncthreads();
    }
    if (tid == 0) sFar = si[0];
    __syncthreads();
  }
}

__global__ void k_gather_centers(const float* pts, const int* fidx, float* ctr) {
  for (long long i = blockIdx.x * (long long)blockDim.x + threadIdx.x;
       i < (long long)PB * PG; i += (long long)gridDim.x * blockDim.x) {
    const int b = (int)(i >> 8);
    const int pI = fidx[i];
#pragma unroll
    for (int j = 0; j < 3; ++j)
      ctr[i * 3 + j] = pts[((long long)b * PN + pI) * 3 + j];
  }
}

// ---------------------------------------------------------------------------
// KNN top-32 by iterative min-selection in LDS: one block per center
// ---------------------------------------------------------------------------
__global__ __launch_bounds__(256) void k_knn(const float* pts, const float* ctr, int* knn) {
  const int cI = blockIdx.x;
  const int b = cI >> 8;
  const float* P = pts + (long long)b * PN * 3;
  const float cx = ctr[cI * 3], cy = ctr[cI * 3 + 1], cz = ctr[cI * 3 + 2];
  __shared__ float sd[PN];
  __shared__ float sv[256];
  __shared__ int   si[256];
  const int tid = threadIdx.x;
  for (int j = 0; j < 32; ++j) {
    const int pI = tid + 256 * j;
    const float dx = P[pI * 3] - cx, dy = P[pI * 3 + 1] - cy, dz = P[pI * 3 + 2] - cz;
    sd[pI] = dx * dx + dy * dy + dz * dz;
  }
  __syncthreads();
  for (int s = 0; s < PK; ++s) {
    float bv = 3.3e38f; int bi = 0;
    for (int j = 0; j < 32; ++j) {
      const int pI = tid + 256 * j;
      const float d = sd[pI];
      if (d < bv) { bv = d; bi = pI; }
    }
    sv[tid] = bv; si[tid] = bi;
    __syncthreads();
    for (int st = 128; st > 0; st >>= 1) {
      if (tid < st) {
        if (sv[tid + st] < sv[tid] ||
            (sv[tid + st] == sv[tid] && si[tid + st] < si[tid])) {
          sv[tid] = sv[tid + st]; si[tid] = si[tid + st];
        }
      }
      __syncthreads();
    }
    if (tid == 0) { knn[(long long)cI * PK + s] = si[0]; sd[si[0]] = 3.4e38f; }
    __syncthreads();
  }
}

__global__ void k_nb(const float* pts, const int* knn, const float* ctr, float* nb) {
  for (long long i = blockIdx.x * (long long)blockDim.x + threadIdx.x;
       i < (long long)PB * PG * PK; i += (long long)gridDim.x * blockDim.x) {
    const int cI = (int)(i >> 5);
    const int b  = (int)(i >> 13);
    const int nbr = knn[i];
    const float* P = pts + (long long)b * PN * 3;
#pragma unroll
    for (int j = 0; j < 3; ++j)
      nb[i * 3 + j] = P[nbr * 3 + j] - ctr[(long long)cI * 3 + j];
  }
}

// ---------------------------------------------------------------------------
// Encoder stage 1: h1 = nb @ e1_w + b  -> global BN stats (recompute) -> bn+relu
// ---------------------------------------------------------------------------
__global__ __launch_bounds__(256) void k_enc1_stats(const float* nb, const float* W,
                                                    const float* bias, float* mv) {
  const int c = blockIdx.x;  // 128 channels
  const int tid = threadIdx.x;
  const long long M = (long long)PB * PG * PK;
  float s = 0.f, sq = 0.f;
  for (long long r = tid; r < M; r += 256) {
    const float h = nb[r * 3] * W[c] + nb[r * 3 + 1] * W[128 + c] +
                    nb[r * 3 + 2] * W[256 + c] + bias[c];
    s += h; sq += h * h;
  }
  __shared__ float rs[256], rq[256];
  rs[tid] = s; rq[tid] = sq;
  __syncthreads();
  for (int st = 128; st > 0; st >>= 1) {
    if (tid < st) { rs[tid] += rs[tid + st]; rq[tid] += rq[tid + st]; }
    __syncthreads();
  }
  if (tid == 0) {
    const float mean = rs[0] / (float)M;
    mv[2 * c] = mean;
    mv[2 * c + 1] = rq[0] / (float)M - mean * mean;
  }
}

__global__ void k_enc1_write(const float* nb, const float* W, const float* bias,
                             const float* mv, const float* g, const float* bb,
                             __bf16* h1n) {
  const long long n = (long long)PB * PG * PK * 128;
  for (long long i = blockIdx.x * (long long)blockDim.x + threadIdx.x;
       i < n; i += (long long)gridDim.x * blockDim.x) {
    const long long r = i >> 7;
    const int c = (int)(i & 127);
    const float h = nb[r * 3] * W[c] + nb[r * 3 + 1] * W[128 + c] +
                    nb[r * 3 + 2] * W[256 + c] + bias[c];
    const float o = (h - mv[2 * c]) * rsqrtf(mv[2 * c + 1] + LEPS) * g[c] + bb[c];
    h1n[i] = f2b(fmaxf(o, 0.f));
  }
}

__global__ __launch_bounds__(256) void k_stats_bf16(const __bf16* x, float* mv,
                                                    long long M, int C) {
  const int c = blockIdx.x;
  const int tid = threadIdx.x;
  float s = 0.f, sq = 0.f;
  for (long long r = tid; r < M; r += 256) {
    const float v = b2f(x[r * C + c]);
    s += v; sq += v * v;
  }
  __shared__ float rs[256], rq[256];
  rs[tid] = s; rq[tid] = sq;
  __syncthreads();
  for (int st = 128; st > 0; st >>= 1) {
    if (tid < st) { rs[tid] += rs[tid + st]; rq[tid] += rq[tid + st]; }
    __syncthreads();
  }
  if (tid == 0) {
    const float mean = rs[0] / (float)M;
    mv[2 * c] = mean;
    mv[2 * c + 1] = rq[0] / (float)M - mean * mean;
  }
}

__global__ void k_bnrelu_bf16(__bf16* x, const float* mv, const float* g,
                              const float* bb, long long n, int C) {
  for (long long i = blockIdx.x * (long long)blockDim.x + threadIdx.x;
       i < n; i += (long long)gridDim.x * blockDim.x) {
    const int c = (int)(i % C);
    float v = b2f(x[i]);
    v = (v - mv[2 * c]) * rsqrtf(mv[2 * c + 1] + LEPS) * g[c] + bb[c];
    x[i] = f2b(fmaxf(v, 0.f));
  }
}

__global__ void k_maxpool_bf16(const __bf16* in, __bf16* out, long long n, int C) {
  for (long long i = blockIdx.x * (long long)blockDim.x + threadIdx.x;
       i < n; i += (long long)gridDim.x * blockDim.x) {
    const long long g = i / C;
    const int c = (int)(i - g * C);
    float m = -3.4e38f;
#pragma unroll
    for (int k = 0; k < PK; ++k)
      m = fmaxf(m, b2f(in[((long long)g * PK + k) * C + c]));
    out[i] = f2b(m);
  }
}

// ---------------------------------------------------------------------------
// misc elementwise
// ---------------------------------------------------------------------------
// transposed convert: src f32 (L, Kd, Nd) row-major -> dst bf16 (L, Nd, Kd)
__global__ void k_cvt_t_bf16(const float* s, __bf16* d, int Kd, int Nd, int L) {
  const long long per = (long long)Kd * Nd;
  const long long n = per * L;
  for (long long i = blockIdx.x * (long long)blockDim.x + threadIdx.x;
       i < n; i += (long long)gridDim.x * blockDim.x) {
    const long long l = i / per;
    const long long rem = i - l * per;
    const int k = (int)(rem / Nd);
    const int nn = (int)(rem - (long long)k * Nd);
    d[l * per + (long long)nn * Kd + k] = f2b(s[i]);
  }
}

__global__ void k_zero_bf16(__bf16* d, long long n) {
  for (long long i = blockIdx.x * (long long)blockDim.x + threadIdx.x;
       i < n; i += (long long)gridDim.x * blockDim.x)
    d[i] = f2b(0.f);
}

__global__ void k_pe1(const float* ctr, const float* W, const float* bias, __bf16* pm) {
  const long long n = (long long)PB * PG * 128;
  for (long long i = blockIdx.x * (long long)blockDim.x + threadIdx.x;
       i < n; i += (long long)gridDim.x * blockDim.x) {
    const long long r = i >> 7;
    const int c = (int)(i & 127);
    const float v = ctr[r * 3] * W[c] + ctr[r * 3 + 1] * W[128 + c] +
                    ctr[r * 3 + 2] * W[256 + c] + bias[c];
    pm[i] = f2b(gelu_f(v));
  }
}

__global__ void k_cls(const float* clsT, const float* clsP, float* x, float* pos) {
  for (long long i = blockIdx.x * (long long)blockDim.x + threadIdx.x;
       i < (long long)PB * EMB; i += (long long)gridDim.x * blockDim.x) {
    const int b = (int)(i / EMB);
    const int c = (int)(i % EMB);
    x[((long long)b * PT) * EMB + c]   = clsT[c];
    pos[((long long)b * PT) * EMB + c] = clsP[c];
  }
}

// ---------------------------------------------------------------------------
// LayerNorm over EMB=384 (optionally adds pos first); bf16 or f32 out
// ---------------------------------------------------------------------------
__global__ __launch_bounds__(128) void k_ln(const float* x, const float* pos,
                                            const float* g, const float* bta,
                                            __bf16* ob, float* of) {
  const int row = blockIdx.x;
  const long long base = (long long)row * EMB;
  const int tid = threadIdx.x;
  __shared__ float red[128];
  float v[3];
#pragma unroll
  for (int j = 0; j < 3; ++j) {
    const int c = tid + j * 128;
    v[j] = x[base + c] + (pos ? pos[base + c] : 0.f);
  }
  red[tid] = v[0] + v[1] + v[2];
  __syncthreads();
  for (int st = 64; st > 0; st >>= 1) {
    if (tid < st) red[tid] += red[tid + st];
    __syncthreads();
  }
  const float mean = red[0] / (float)EMB;
  __syncthreads();
  const float d0 = v[0] - mean, d1 = v[1] - mean, d2 = v[2] - mean;
  red[tid] = d0 * d0 + d1 * d1 + d2 * d2;
  __syncthreads();
  for (int st = 64; st > 0; st >>= 1) {
    if (tid < st) red[tid] += red[tid + st];
    __syncthreads();
  }
  const float rs = rsqrtf(red[0] / (float)EMB + LEPS);
#pragma unroll
  for (int j = 0; j < 3; ++j) {
    const int c = tid + j * 128;
    const float o = (v[j] - mean) * rs * g[c] + bta[c];
    if (ob) ob[base + c] = f2b(o);
    else    of[base + c] = o;
  }
}

// masked, scaled softmax over padded rows; writes bf16 probs (pad == 0)
__global__ __launch_bounds__(128) void k_softmax(const float* sc, __bf16* pr) {
  const long long rg = blockIdx.x;
  const int row = (int)(rg % TPAD);
  const float* s = sc + rg * TPAD;
  __bf16* p = pr + rg * TPAD;
  const int tid = threadIdx.x;
  if (row >= PT) {
    for (int c = tid; c < TPAD; c += 128) p[c] = f2b(0.f);
    return;
  }
  __shared__ float red[128];
  float mx = -3.4e38f;
  for (int c = tid; c < PT; c += 128) mx = fmaxf(mx, s[c] * 0.125f);
  red[tid] = mx;
  __syncthreads();
  for (int st = 64; st > 0; st >>= 1) {
    if (tid < st) red[tid] = fmaxf(red[tid], red[tid + st]);
    __syncthreads();
  }
  mx = red[0];
  __syncthreads();
  float sum = 0.f;
  for (int c = tid; c < PT; c += 128) sum += expf(s[c] * 0.125f - mx);
  red[tid] = sum;
  __syncthreads();
  for (int st = 64; st > 0; st >>= 1) {
    if (tid < st) red[tid] += red[tid + st];
    __syncthreads();
  }
  const float inv = 1.f / red[0];
  for (int c = tid; c < TPAD; c += 128)
    p[c] = (c < PT) ? f2b(expf(s[c] * 0.125f - mx) * inv) : f2b(0.f);
}

__global__ void k_feat(const float* xn, float* feat) {
  for (long long i = blockIdx.x * (long long)blockDim.x + threadIdx.x;
       i < (long long)PB * EMB; i += (long long)gridDim.x * blockDim.x) {
    const int b = (int)(i / EMB);
    const int c = (int)(i % EMB);
    feat[(long long)b * 2 * EMB + c] = xn[((long long)b * PT) * EMB + c];
    float m = -3.4e38f;
    for (int t = 1; t < PT; ++t)
      m = fmaxf(m, xn[((long long)b * PT + t) * EMB + c]);
    feat[(long long)b * 2 * EMB + EMB + c] = m;
  }
}

__global__ void k_head_gemm(const float* A, const float* W, const float* bias,
                            float* out, int M, int Kd, int Nd) {
  for (long long i = blockIdx.x * (long long)blockDim.x + threadIdx.x;
       i < (long long)M * Nd; i += (long long)gridDim.x * blockDim.x) {
    const int r = (int)(i / Nd);
    const int c = (int)(i % Nd);
    float s = bias[c];
    for (int k = 0; k < Kd; ++k)
      s += A[(long long)r * Kd + k] * W[(long long)k * Nd + c];
    out[i] = s;
  }
}

// head BN (32 rows) + relu, one wave per channel
__global__ __launch_bounds__(32) void k_head_bn(float* x, const float* g,
                                                const float* bta, int C) {
  const int c = blockIdx.x;
  const int t = threadIdx.x;
  const float v = x[(long long)t * C + c];
  float s = v, sq = v * v;
  for (int m = 16; m > 0; m >>= 1) {
    s  += __shfl_xor(s, m, 32);
    sq += __shfl_xor(sq, m, 32);
  }
  const float mean = s / 32.f;
  const float var  = sq / 32.f - mean * mean;
  const float o = (v - mean) * rsqrtf(var + LEPS) * g[c] + bta[c];
  x[(long long)t * C + c] = fmaxf(o, 0.f);
}

// ---------------------------------------------------------------------------
// host side
// ---------------------------------------------------------------------------
static inline int gsz(long long n) {
  long long b = (n + 255) / 256;
  if (b > 262144) b = 262144;
  if (b < 1) b = 1;
  return (int)b;
}

template <int AMODE, int EPI, int ACT, int NT>
static void launch_gemm(hipStream_t st, const void* A, const void* B, const float* bias,
                        void* out, const float* R1, const float* R2, const void* Afg,
                        void* q, void* k2, void* v,
                        int M, int N, int K, int lda, int ldb, int ldc,
                        long long sA, long long sB, long long sC, int nbatch) {
  GemmP p;
  p.A = (const __bf16*)A; p.B = (const __bf16*)B; p.bias = bias;
  p.out = out; p.R1 = R1; p.R2 = R2; p.Afg = (const __bf16*)Afg;
  p.q = (__bf16*)q; p.k2 = (__bf16*)k2; p.v = (__bf16*)v;
  p.strideA = sA; p.strideB = sB; p.strideC = sC;
  p.M = M; p.N = N; p.K = K; p.lda = lda; p.ldb = ldb; p.ldc = ldc;
  const int tiles = (M / 16) * (N / (16 * NT));
  dim3 grid((tiles + 7) / 8, nbatch);
  k_gemm<AMODE, EPI, ACT, NT><<<grid, 256, 0, st>>>(p);
}

extern "C" void kernel_launch(void* const* d_in, const int* in_sizes, int n_in,
                              void* d_out, int out_size, void* d_ws, size_t ws_size,
                              hipStream_t stream) {
  (void)in_sizes; (void)n_in; (void)out_size; (void)ws_size;
  const float* pts    = (const float*)d_in[0];
  const float* e1_w   = (const float*)d_in[1];
  const float* e1_b   = (const float*)d_in[2];
  const float* e1bn_g = (const float*)d_in[3];
  const float* e1bn_b = (const float*)d_in[4];
  const float* e2_w   = (const float*)d_in[5];
  const float* e2_b   = (const float*)d_in[6];
  const float* e3_w   = (const float*)d_in[7];
  const float* e3_b   = (const float*)d_in[8];
  const float* e3bn_g = (const float*)d_in[9];
  const float* e3bn_b = (const float*)d_in[10];
  const float* e4_w   = (const float*)d_in[11];
  const float* e4_b   = (const float*)d_in[12];
  const float* red_w  = (const float*)d_in[13];
  const float* red_b  = (const float*)d_in[14];
  const float* clsT   = (const float*)d_in[15];
  const float* clsP   = (const float*)d_in[16];
  const float* pe1_w  = (const float*)d_in[17];
  const float* pe1_b  = (const float*)d_in[18];
  const float* pe2_w  = (const float*)d_in[19];
  const float* pe2_b  = (const float*)d_in[20];
  const float* ln1_g  = (const float*)d_in[21];
  const float* ln1_b  = (const float*)d_in[22];
  const float* qkv_w  = (const float*)d_in[23];
  const float* proj_w = (const float*)d_in[24];
  const float* proj_b = (const float*)d_in[25];
  const float* ln2_g  = (const float*)d_in[26];
  const float* ln2_b  = (const float*)d_in[27];
  const float* m1_w   = (const float*)d_in[28];
  const float* m1_b   = (const float*)d_in[29];
  const float* m2_w   = (const float*)d_in[30];
  const float* m2_b   = (const float*)d_in[31];
  const float* ln_g   = (const float*)d_in[32];
  const float* ln_b   = (const float*)d_in[33];
  const float* h1_w   = (const float*)d_in[34];
  const float* h1_b   = (const float*)d_in[35];
  const float* h1bn_g = (const float*)d_in[36];
  const float* h1bn_b = (const float*)d_in[37];
  const float* h2_w   = (const float*)d_in[38];
  const float* h2_b   = (const float*)d_in[39];
  const float* h2bn_g = (const float*)d_in[40];
  const float* h2bn_b = (const float*)d_in[41];
  const float* h3_w   = (const float*)d_in[42];
  const float* h3_b   = (const float*)d_in[43];

  char* w = (char*)d_ws;
  size_t off = 0;
  auto take = [&](size_t bytes) -> char* {
    char* r = w + off;
    off = (off + bytes + 255) & ~(size_t)255;
    return r;
  };

  const long long MPTS = (long long)PB * PG * PK;      // 262144
  const long long ROWS = (long long)PB * PT;           // 8224
  const long long GRPS = (long long)PB * PG;           // 8192

  int*    fpsIdx = (int*)take(GRPS * 4);
  float*  ctr    = (float*)take(GRPS * 3 * 4);
  int*    knn    = (int*)take(MPTS * 4);
  float*  nb     = (float*)take(MPTS * 3 * 4);
  float*  mv     = (float*)take(512 * 2 * 4);
  __bf16* fg     = (__bf16*)take(GRPS * ENCD * 2);
  __bf16* tok    = (__bf16*)take(GRPS * ENCD * 2);
  __bf16* pm     = (__bf16*)take(GRPS * 128 * 2);
  __bf16* e2wb   = (__bf16*)take(128 * 256 * 2);
  __bf16* e3wb   = (__bf16*)take(512 * 512 * 2);
  __bf16* e4wb   = (__bf16*)take(512 * 256 * 2);
  __bf16* redwb  = (__bf16*)take(256 * 384 * 2);
  __bf16* pe2wb  = (__bf16*)take(128 * 384 * 2);
  __bf16* qkvwb  = (__bf16*)take((size_t)NLAYER * 384 * 1152 * 2);
  __bf16* projwb = (__bf16*)take((size_t)NLAYER * 384 * 384 * 2);
  __bf16* m1wb   = (__bf16*)take((size_t)NLAYER * 384 * 1536 * 2);
  __bf16* m2wb   = (__bf16*)take((size_t)NLAYER * 1536 * 384 * 2);
  float*  posb   = (float*)take(ROWS * EMB * 4);
  float*  xbuf   = (float*)take(ROWS * EMB * 4);
  float*  x1buf  = (float*)take(ROWS * EMB * 4);
  __bf16* hbf    = (__bf16*)take(ROWS * EMB * 2);
  __bf16* obf    = (__bf16*)take(ROWS * EMB * 2);
  float*  feat   = (float*)take((size_t)PB * 2 * EMB * 4);
  float*  g1     = (float*)take((size_t)PB * 256 * 4);
  float*  g2     = (float*)take((size_t)PB * 256 * 4);

  // shared arena: encoder scratch vs transformer scratch (disjoint lifetimes)
  const size_t encBytes = (size_t)MPTS * 128 * 2 + (size_t)MPTS * 256 * 2 +
                          (size_t)MPTS * 512 * 2 + (size_t)MPTS * 256 * 2;
  char* arena = take(encBytes);
  __bf16* h1n = (__bf16*)arena;
  __bf16* f2  = (__bf16*)(arena + (size_t)MPTS * 128 * 2);
  __bf16* h3  = (__bf16*)(arena + (size_t)MPTS * 128 * 2 + (size_t)MPTS * 256 * 2);
  __bf16* f4  = (__bf16*)(arena + (size_t)MPTS * 128 * 2 + (size_t)MPTS * 256 * 2 +
                          (size_t)MPTS * 512 * 2);
  const long long ZB = (long long)PB * NH;                  // 192
  const long long qkvN = ZB * TPAD * HDIM;                  // 3538944
  __bf16* Qp    = (__bf16*)arena;
  __bf16* Kp    = Qp + qkvN;
  __bf16* Vp    = Kp + qkvN;       // stored transposed: (B*H, HDIM, TPAD)
  float*  score = (float*)(Vp + qkvN);
  __bf16* probs = (__bf16*)((char*)score + (size_t)ZB * TPAD * TPAD * 4);
  __bf16* mlph  = probs + (size_t)ZB * TPAD * TPAD;

  // ---- grouping: FPS -> centers -> KNN -> relative neighborhoods ----
  k_fps<<<PB, 256, 0, stream>>>(pts, fpsIdx);
  k_gather_centers<<<gsz(GRPS), 256, 0, stream>>>(pts, fpsIdx, ctr);
  k_knn<<<(int)GRPS, 256, 0, stream>>>(pts, ctr, knn);
  k_nb<<<gsz(MPTS), 256, 0, stream>>>(pts, knn, ctr, nb);

  // ---- weight conversion to transposed bf16 (B^T, N x K row-major) ----
  k_cvt_t_bf16<<<gsz(128 * 256), 256, 0, stream>>>(e2_w, e2wb, 128, 256, 1);
  k_cvt_t_bf16<<<gsz(512 * 512), 256, 0, stream>>>(e3_w, e3wb, 512, 512, 1);
  k_cvt_t_bf16<<<gsz(512 * 256), 256, 0, stream>>>(e4_w, e4wb, 512, 256, 1);
  k_cvt_t_bf16<<<gsz(256 * 384), 256, 0, stream>>>(red_w, redwb, 256, 384, 1);
  k_cvt_t_bf16<<<gsz(128 * 384), 256, 0, stream>>>(pe2_w, pe2wb, 128, 384, 1);
  k_cvt_t_bf16<<<gsz((long long)NLAYER * 442368), 256, 0, stream>>>(qkv_w, qkvwb, 384, 1152, NLAYER);
  k_cvt_t_bf16<<<gsz((long long)NLAYER * 147456), 256, 0, stream>>>(proj_w, projwb, 384, 384, NLAYER);
  k_cvt_t_bf16<<<gsz((long long)NLAYER * 589824), 256, 0, stream>>>(m1_w, m1wb, 384, 1536, NLAYER);
  k_cvt_t_bf16<<<gsz((long long)NLAYER * 589824), 256, 0, stream>>>(m2_w, m2wb, 1536, 384, NLAYER);

  // ---- encoder ----
  k_enc1_stats<<<128, 256, 0, stream>>>(nb, e1_w, e1_b, mv);
  k_enc1_write<<<gsz(MPTS * 128), 256, 0, stream>>>(nb, e1_w, e1_b, mv, e1bn_g, e1bn_b, h1n);
  launch_gemm<0, 1, 0, 4>(stream, h1n, e2wb, e2_b, f2, nullptr, nullptr, nullptr,
                          nullptr, nullptr, nullptr,
                          (int)MPTS, 256, 128, 128, 128, 256, 0, 0, 0, 1);
  k_maxpool_bf16<<<gsz(GRPS * ENCD), 256, 0, stream>>>(f2, fg, GRPS * ENCD, ENCD);
  launch_gemm<1, 1, 0, 4>(stream, f2, e3wb, e3_b, h3, nullptr, nullptr, fg,
                          nullptr, nullptr, nullptr,
                          (int)MPTS, 512, 512, 0, 512, 512, 0, 0, 0, 1);
  k_stats_bf16<<<512, 256, 0, stream>>>(h3, mv, MPTS, 512);
  k_bnrelu_bf16<<<gsz(MPTS * 512), 256, 0, stream>>>(h3, mv, e3bn_g, e3bn_b, MPTS * 512, 512);
  launch_gemm<0, 1, 0, 4>(stream, h3, e4wb, e4_b, f4, nullptr, nullptr, nullptr,
                          nullptr, nullptr, nullptr,
                          (int)MPTS, 256, 512, 512, 512, 256, 0, 0, 0, 1);
  k_maxpool_bf16<<<gsz(GRPS * ENCD), 256, 0, stream>>>(f4, tok, GRPS * ENCD, ENCD);

  // ---- embedding: token reduce + pos MLP + cls slots ----
  launch_gemm<0, 4, 0, 4>(stream, tok, redwb, red_b, xbuf, nullptr, nullptr, nullptr,
                          nullptr, nullptr, nullptr,
                          (int)GRPS, EMB, 256, 256, 256, EMB, 0, 0, 0, 1);
  k_pe1<<<gsz(GRPS * 128), 256, 0, stream>>>(ctr, pe1_w, pe1_b, pm);
  launch_gemm<0, 4, 0, 4>(stream, pm, pe2wb, pe2_b, posb, nullptr, nullptr, nullptr,
                          nullptr, nullptr, nullptr,
                          (int)GRPS, EMB, 128, 128, 128, EMB, 0, 0, 0, 1);
  k_cls<<<gsz((long long)PB * EMB), 256, 0, stream>>>(clsT, clsP, xbuf, posb);

  // zero-pad Q/K/V once (valid regions rewritten every layer)
  k_zero_bf16<<<gsz(qkvN), 256, 0, stream>>>(Qp, qkvN);
  k_zero_bf16<<<gsz(qkvN), 256, 0, stream>>>(Kp, qkvN);
  k_zero_bf16<<<gsz(qkvN), 256, 0, stream>>>(Vp, qkvN);

  // ---- transformer ----
  for (int l = 0; l < NLAYER; ++l) {
    k_ln<<<(int)ROWS, 128, 0, stream>>>(xbuf, posb, ln1_g + l * EMB, ln1_b + l * EMB,
                                        hbf, nullptr);
    launch_gemm<0, 2, 0, 4>(stream, hbf, qkvwb + (size_t)l * 442368, nullptr, nullptr,
                            nullptr, nullptr, nullptr, Qp, Kp, Vp,
                            (int)ROWS, 3 * EMB, EMB, EMB, EMB, 0, 0, 0, 0, 1);
    launch_gemm<0, 0, 0, 2>(stream, Qp, Kp, nullptr, score, nullptr, nullptr, nullptr,
                            nullptr, nullptr, nullptr,
                            TPAD, TPAD, HDIM, HDIM, HDIM, TPAD,
                            (long long)TPAD * HDIM, (long long)TPAD * HDIM,
                            (long long)TPAD * TPAD, (int)ZB);
    k_softmax<<<(int)(ZB * TPAD), 128, 0, stream>>>(score, probs);
    launch_gemm<0, 3, 0, 4>(stream, probs, Vp, nullptr, obf, nullptr, nullptr, nullptr,
                            nullptr, nullptr, nullptr,
                            TPAD, HDIM, TPAD, TPAD, TPAD, HDIM,
                            (long long)TPAD * TPAD, (long long)TPAD * HDIM, 0, (int)ZB);
    launch_gemm<0, 5, 0, 4>(stream, obf, projwb + (size_t)l * 147456, proj_b + l * EMB,
                            x1buf, xbuf, posb, nullptr, nullptr, nullptr, nullptr,
                            (int)ROWS, EMB, EMB, EMB, EMB, EMB, 0, 0, 0, 1);
    k_ln<<<(int)ROWS, 128, 0, stream>>>(x1buf, nullptr, ln2_g + l * EMB, ln2_b + l * EMB,
                                        hbf, nullptr);
    launch_gemm<0, 1, 2, 4>(stream, hbf, m1wb + (size_t)l * 589824, m1_b + l * 4 * EMB,
                            mlph, nullptr, nullptr, nullptr, nullptr, nullptr, nullptr,
                            (int)ROWS, 4 * EMB, EMB, EMB, EMB, 4 * EMB, 0, 0, 0, 1);
    launch_gemm<0, 5, 0, 4>(stream, mlph, m2wb + (size_t)l * 589824, m2_b + l * EMB,
                            xbuf, x1buf, nullptr, nullptr, nullptr, nullptr, nullptr,
                            (int)ROWS, EMB, 4 * EMB, 4 * EMB, 4 * EMB, EMB, 0, 0, 0, 1);
  }

  // ---- head ----
  k_ln<<<(int)ROWS, 128, 0, stream>>>(xbuf, nullptr, ln_g, ln_b, nullptr, x1buf);
  k_feat<<<gsz((long long)PB * EMB), 256, 0, stream>>>(x1buf, feat);
  k_head_gemm<<<gsz((long long)PB * 256), 256, 0, stream>>>(feat, h1_w, h1_b, g1,
                                                            PB, 2 * EMB, 256);
  k_head_bn<<<256, 32, 0, stream>>>(g1, h1bn_g, h1bn_b, 256);
  k_head_gemm<<<gsz((long long)PB * 256), 256, 0, stream>>>(g1, h2_w, h2_b, g2,
                                                            PB, 256, 256);
  k_head_bn<<<256, 32, 0, stream>>>(g2, h2bn_g, h2bn_b, 256);
  k_head_gemm<<<gsz((long long)PB * 40), 256, 0, stream>>>(g2, h3_w, h3_b,
                                                           (float*)d_out, PB, 256, 40);
}
